// Attention_26079041421920
// MI455X (gfx1250) — compile-verified, matrix operand order checked by reading, not run on hardware
//
#include <hip/hip_runtime.h>

// Problem constants (from the reference): B=4, T=2048, C=H=1024
#define B_ 4
#define T_ 2048
#define C_ 1024
#define H_ 1024
#define M_ (B_ * T_)   // 8192 total rows

typedef __bf16 bf16;
typedef __attribute__((ext_vector_type(16))) bf16  v16bf;
typedef __attribute__((ext_vector_type(8)))  bf16  v8bf;
typedef __attribute__((ext_vector_type(8)))  float v8f;

// ---- helpers -------------------------------------------------------------

__device__ __forceinline__ bf16 f2bf(float f) {
    unsigned u = __builtin_bit_cast(unsigned, f);
    u += 0x7FFFu + ((u >> 16) & 1u);            // round-to-nearest-even
    unsigned short h = (unsigned short)(u >> 16);
    return __builtin_bit_cast(bf16, h);
}

__device__ __forceinline__ v8f vzero8() {
    v8f z;
#pragma unroll
    for (int i = 0; i < 8; ++i) z[i] = 0.0f;
    return z;
}

// Load a 16x32 bf16 WMMA fragment from a row-major tile (row stride `ld`
// elements). CDNA5 layout: lane holds row r = lane%16; element j maps to
// k = (lane/16)*8 + j (j<8) and k = 16 + (lane/16)*8 + (j-8). A-frags and
// B-frags (implicitly transposed operand) use the identical loader.
__device__ __forceinline__ v16bf load_frag(const bf16* __restrict__ base,
                                           int ld, int lane) {
    int r    = lane & 15;
    int half = lane >> 4;
    const bf16* p = base + (size_t)r * ld + half * 8;
    v8bf lo = *(const v8bf*)(p);        // k = half*8 + 0..7
    v8bf hi = *(const v8bf*)(p + 16);   // k = 16 + half*8 + 0..7
    return __builtin_shufflevector(lo, hi,
        0, 1, 2, 3, 4, 5, 6, 7, 8, 9, 10, 11, 12, 13, 14, 15);
}

__device__ __forceinline__ v8f wmma_bf16(v16bf a, v16bf b, v8f c) {
    return __builtin_amdgcn_wmma_f32_16x16x32_bf16(
        /*neg_a=*/false, a, /*neg_b=*/false, b,
        /*c_mod=*/(short)0, c, /*reuse_a=*/false, /*reuse_b=*/false);
}

// Async global->LDS copy (GLOBAL_LOAD_ASYNC_TO_LDS_B128, GV mode).
// Generic pointers to __shared__ carry the LDS byte offset in their low 32
// bits (ISA aperture rule), so VDST is just the truncated pointer.
__device__ __forceinline__ void async_b128(unsigned lds_addr, const void* g) {
    unsigned long long ga = (unsigned long long)(uintptr_t)g;
    asm volatile("global_load_async_to_lds_b128 %0, %1, off"
                 :: "v"(lds_addr), "v"(ga) : "memory");
}
__device__ __forceinline__ void wait_async_le4() {
    asm volatile("s_wait_asynccnt 4" ::: "memory");
}
__device__ __forceinline__ void wait_async_0() {
    asm volatile("s_wait_asynccnt 0" ::: "memory");
}

// ---- kernel 1: fp32 -> bf16 convert (optionally transposed, scaled) ------

__global__ __launch_bounds__(256)
void convert_kernel(const float* __restrict__ src, bf16* __restrict__ dst,
                    int rows, int cols, int transpose, float scale) {
    int idx = blockIdx.x * 256 + threadIdx.x;
    if (idx >= rows * cols) return;
    bf16 v = f2bf(src[idx] * scale);
    if (!transpose) {
        dst[idx] = v;
    } else {
        int k = idx / cols;                     // source row
        int n = idx - k * cols;                 // source col
        dst[(size_t)n * rows + k] = v;          // dst[n][k]
    }
}

// ---- kernel 2: LDS-tiled bf16 GEMM  D[m][n] = sum_k A[m][k]*Wt[n][k] -----
// 256 threads / 8 waves per 128x128 tile. A/B k-slices (128x32 bf16) are
// double-buffered in LDS via async global->LDS copies (ASYNCcnt), rows
// padded to 80B to spread DS banks. Wave w computes a 32x64 sub-tile
// (2x4 WMMA tiles): 8 WMMAs per 16KB staged -> 64 flop/B from L2.

#define LDA 40   // padded LDS row stride in bf16 elements (80 bytes)

__global__ __launch_bounds__(256)
void gemm_qkv_kernel(const bf16* __restrict__ A, const bf16* __restrict__ Wt,
                     bf16* __restrict__ D, int storeT) {
    __shared__ __align__(16) bf16 Abuf[2][128][LDA];
    __shared__ __align__(16) bf16 Bbuf[2][128][LDA];

    int tid  = threadIdx.x;
    int lane = tid & 31;
    int wave = tid >> 5;
    int n0 = blockIdx.x * 128;
    int m0 = blockIdx.y * 128;
    int msub = (wave & 3) * 32;                 // 4 m-waves
    int nsub = (wave >> 2) * 64;                // 2 n-waves

    // this thread's two 16B staging chunks: chunk -> (row, 16B col)
    int ch0 = tid, ch1 = tid + 256;             // 512 chunks = 128 rows x 4
    int r0 = ch0 >> 2, c0 = ch0 & 3;
    int r1 = ch1 >> 2, c1 = ch1 & 3;

    const bf16* Ab = A  + (size_t)m0 * C_;
    const bf16* Bb = Wt + (size_t)n0 * C_;

    v8f acc[2][4];
#pragma unroll
    for (int i = 0; i < 2; ++i)
#pragma unroll
        for (int j = 0; j < 4; ++j) acc[i][j] = vzero8();

    // prologue: stage k-slice 0 into buffer 0
    {
        unsigned la = (unsigned)(uintptr_t)&Abuf[0][0][0];
        unsigned lb = (unsigned)(uintptr_t)&Bbuf[0][0][0];
        async_b128(la + r0 * 80 + c0 * 16, Ab + (size_t)r0 * C_ + c0 * 8);
        async_b128(la + r1 * 80 + c1 * 16, Ab + (size_t)r1 * C_ + c1 * 8);
        async_b128(lb + r0 * 80 + c0 * 16, Bb + (size_t)r0 * C_ + c0 * 8);
        async_b128(lb + r1 * 80 + c1 * 16, Bb + (size_t)r1 * C_ + c1 * 8);
    }

    const int NK = C_ / 32;                     // 32 k-steps
    for (int kt = 0; kt < NK; ++kt) {
        int cur = kt & 1;
        if (kt + 1 < NK) {                      // stage next slice
            int k0 = (kt + 1) * 32;
            unsigned la = (unsigned)(uintptr_t)&Abuf[cur ^ 1][0][0];
            unsigned lb = (unsigned)(uintptr_t)&Bbuf[cur ^ 1][0][0];
            async_b128(la + r0 * 80 + c0 * 16, Ab + (size_t)r0 * C_ + k0 + c0 * 8);
            async_b128(la + r1 * 80 + c1 * 16, Ab + (size_t)r1 * C_ + k0 + c1 * 8);
            async_b128(lb + r0 * 80 + c0 * 16, Bb + (size_t)r0 * C_ + k0 + c0 * 8);
            async_b128(lb + r1 * 80 + c1 * 16, Bb + (size_t)r1 * C_ + k0 + c1 * 8);
            wait_async_le4();                   // current slice (prev 4) done
        } else {
            wait_async_0();
        }
        __syncthreads();                        // all waves staged `cur`

        v16bf a0 = load_frag(&Abuf[cur][msub     ][0], LDA, lane);
        v16bf a1 = load_frag(&Abuf[cur][msub + 16][0], LDA, lane);
#pragma unroll
        for (int j = 0; j < 4; ++j) {
            v16bf bj = load_frag(&Bbuf[cur][nsub + j * 16][0], LDA, lane);
            acc[0][j] = wmma_bf16(a0, bj, acc[0][j]);
            acc[1][j] = wmma_bf16(a1, bj, acc[1][j]);
        }
        __syncthreads();                        // done reading before overwrite
    }

    // C/D layout: lane -> n = lane%16; VGPR i -> m = i + 8*(lane/16)
    int nl = lane & 15;
    int mh = (lane >> 4) * 8;
#pragma unroll
    for (int i2 = 0; i2 < 2; ++i2) {
#pragma unroll
        for (int j = 0; j < 4; ++j) {
            int n = n0 + nsub + j * 16 + nl;
#pragma unroll
            for (int i = 0; i < 8; ++i) {
                int m = m0 + msub + i2 * 16 + mh + i;
                bf16 v = f2bf(acc[i2][j][i]);
                if (!storeT) {
                    D[(size_t)m * H_ + n] = v;
                } else {                         // V transposed: Vt[b][h][t]
                    int b = m >> 11;             // m / T_
                    int t = m & (T_ - 1);
                    D[((size_t)b * H_ + n) * T_ + t] = v;
                }
            }
        }
    }
}

// ---- kernel 3: fused causal flash attention ------------------------------
// Workgroup = 8 waves handles 16 query rows; wave w owns h-slice
// [w*128, w*128+128). Score tiles (16x32 over full h=1024) are reduced
// across waves via LDS float atomics; online softmax in fp32; P*V
// accumulated with WMMA into per-wave fp32 accumulators.

__global__ __launch_bounds__(256)
void attn_kernel(const bf16* __restrict__ Qb, const bf16* __restrict__ Kb,
                 const bf16* __restrict__ Vt, float* __restrict__ Out) {
    __shared__ __align__(16) float S_red[16][32];
    __shared__ __align__(16) bf16  Ptile[16][32];
    __shared__ float mstate[16], lstate[16], alpha_s[16], mnew_s[16];

    int tid  = threadIdx.x;
    int lane = tid & 31;
    int wave = tid >> 5;
    int b    = blockIdx.y;
    int t0   = blockIdx.x * 16;
    int hbase = wave * 128;

    const bf16* Qbb = Qb + ((size_t)b * T_ + t0) * H_;
    const bf16* Kbb = Kb + (size_t)b * T_ * H_;
    const bf16* Vtb = Vt + (size_t)b * H_ * T_;

    // Q fragments for this wave's 128-wide h slice (scale already folded in)
    v16bf qf[4];
#pragma unroll
    for (int kk = 0; kk < 4; ++kk)
        qf[kk] = load_frag(Qbb + hbase + kk * 32, H_, lane);

    v8f acc[8];                                  // O tile: 16 x 128 per wave
#pragma unroll
    for (int j = 0; j < 8; ++j) acc[j] = vzero8();

    if (tid < 16) { mstate[tid] = -1e30f; lstate[tid] = 0.0f; }

    int nsteps = (t0 + 16 + 31) >> 5;            // causal: s <= t0+15
    for (int step = 0; step < nsteps; ++step) {
        int s0 = step << 5;

        ((float*)S_red)[tid]       = 0.0f;       // zero 512 floats
        ((float*)S_red)[tid + 256] = 0.0f;
        __syncthreads();

        // Partial scores: S(16x32) += Qslice(16x128) * Kslice^T(128x32)
#pragma unroll
        for (int half = 0; half < 2; ++half) {
            v8f sa = vzero8();
#pragma unroll
            for (int kk = 0; kk < 4; ++kk) {
                v16bf kf = load_frag(
                    Kbb + (size_t)(s0 + half * 16) * H_ + hbase + kk * 32,
                    H_, lane);
                sa = wmma_bf16(qf[kk], kf, sa);
            }
            int nl = lane & 15, mh = (lane >> 4) * 8;
#pragma unroll
            for (int i = 0; i < 8; ++i)
                atomicAdd(&S_red[mh + i][half * 16 + nl], sa[i]);  // ds_add_f32
        }
        __syncthreads();

        // Online softmax statistics (one thread per query row)
        if (tid < 16) {
            int   t  = t0 + tid;
            float mo = mstate[tid];
            float mx = mo;
#pragma unroll
            for (int c = 0; c < 32; ++c) {
                float v = (s0 + c <= t) ? S_red[tid][c] : -1e30f;
                mx = fmaxf(mx, v);
            }
            float al  = __expf(mo - mx);
            float sum = 0.0f;
#pragma unroll
            for (int c = 0; c < 32; ++c) {
                float v = (s0 + c <= t) ? S_red[tid][c] : -1e30f;
                sum += __expf(v - mx);
            }
            lstate[tid]  = lstate[tid] * al + sum;
            mstate[tid]  = mx;
            alpha_s[tid] = al;
            mnew_s[tid]  = mx;
        }
        __syncthreads();

        // P tile (bf16) into LDS
        for (int idx = tid; idx < 512; idx += 256) {
            int r = idx >> 5, c = idx & 31;
            float v = (s0 + c <= t0 + r) ? S_red[r][c] : -1e30f;
            Ptile[r][c] = f2bf(__expf(v - mnew_s[r]));
        }
        __syncthreads();

        // Rescale O by alpha, then O += P * V  (B-frags from transposed V)
        {
            int mh = (lane >> 4) * 8;
            float al[8];
#pragma unroll
            for (int i = 0; i < 8; ++i) al[i] = alpha_s[mh + i];
            v16bf pf = load_frag(&Ptile[0][0], 32, lane);
#pragma unroll
            for (int j = 0; j < 8; ++j) {
#pragma unroll
                for (int i = 0; i < 8; ++i) acc[j][i] *= al[i];
                v16bf vf = load_frag(
                    Vtb + (size_t)(hbase + j * 16) * T_ + s0, T_, lane);
                acc[j] = wmma_bf16(pf, vf, acc[j]);
            }
        }
        __syncthreads();
    }

    // Normalize by the softmax denominator and write fp32 output
    int nl = lane & 15, mh = (lane >> 4) * 8;
    float linv[8];
#pragma unroll
    for (int i = 0; i < 8; ++i) linv[i] = 1.0f / lstate[mh + i];
    float* Ob = Out + (size_t)b * T_ * H_;
#pragma unroll
    for (int j = 0; j < 8; ++j) {
        int n = hbase + j * 16 + nl;
#pragma unroll
        for (int i = 0; i < 8; ++i)
            Ob[(size_t)(t0 + mh + i) * H_ + n] = acc[j][i] * linv[i];
    }
}

// ---- host-side launcher --------------------------------------------------

extern "C" void kernel_launch(void* const* d_in, const int* in_sizes, int n_in,
                              void* d_out, int out_size, void* d_ws, size_t ws_size,
                              hipStream_t stream) {
    (void)in_sizes; (void)n_in; (void)out_size; (void)ws_size;
    const float* x  = (const float*)d_in[0];
    const float* Wk = (const float*)d_in[1];
    const float* Wq = (const float*)d_in[2];
    const float* Wv = (const float*)d_in[3];
    float* out = (float*)d_out;

    // Workspace layout (bf16), ~70 MB total
    bf16* xb  = (bf16*)d_ws;                      // [M_][C_]
    bf16* Wtk = xb  + (size_t)M_ * C_;            // [H_][C_]  (transposed)
    bf16* Wtq = Wtk + (size_t)C_ * H_;
    bf16* Wtv = Wtq + (size_t)C_ * H_;
    bf16* Qb  = Wtv + (size_t)C_ * H_;            // [M_][H_]
    bf16* Kb  = Qb  + (size_t)M_ * H_;            // [M_][H_]
    bf16* Vt  = Kb  + (size_t)M_ * H_;            // [B_][H_][T_] (transposed)

    int nx = M_ * C_;
    int nw = C_ * H_;
    convert_kernel<<<(nx + 255) / 256, 256, 0, stream>>>(x,  xb,  M_, C_, 0, 1.0f);
    convert_kernel<<<(nw + 255) / 256, 256, 0, stream>>>(Wk, Wtk, C_, H_, 1, 1.0f);
    // fold softmax scale 1/sqrt(H) = 1/32 into Wq (exact power of two)
    convert_kernel<<<(nw + 255) / 256, 256, 0, stream>>>(Wq, Wtq, C_, H_, 1, 0.03125f);
    convert_kernel<<<(nw + 255) / 256, 256, 0, stream>>>(Wv, Wtv, C_, H_, 1, 1.0f);

    dim3 ggrid(H_ / 128, M_ / 128);               // (8, 64) 128x128 tiles
    gemm_qkv_kernel<<<ggrid, 256, 0, stream>>>(xb, Wtq, Qb, 0);
    gemm_qkv_kernel<<<ggrid, 256, 0, stream>>>(xb, Wtk, Kb, 0);
    gemm_qkv_kernel<<<ggrid, 256, 0, stream>>>(xb, Wtv, Vt, 1);

    dim3 agrid(T_ / 16, B_);                      // (128, 4), 256 threads
    attn_kernel<<<agrid, 256, 0, stream>>>(Qb, Kb, Vt, out);
}
